// Attention_84078279786850
// MI455X (gfx1250) — compile-verified, matrix operand order checked by reading, not run on hardware
//
#include <hip/hip_runtime.h>
#include <math.h>

typedef __attribute__((ext_vector_type(16))) __bf16 v16bf;
typedef __attribute__((ext_vector_type(8)))  float  v8f;

#define WINDOW 128
#define NHEADS 16
#define SEQ    16384
#define DHEAD  32
#define NWIN   (SEQ / WINDOW)      // 128
#define NMEM   4
#define NKEYS  (NMEM + 2*WINDOW)   // 260
#define NKPAD  288                 // 9 key tiles of 32
#define NKT    (NKPAD / 32)        // 9
#define SOFTCLAMP 50.0f
#define NEGINF   (-1e30f)

// ---- LDS layout (bytes). Strides are multiples of 16B (b128 alignment) and
// give conflict-free banks for the 16-lane fragment reads.
#define KROW_STRIDE 80                         // 64B data + 16B pad
#define K_OFF   0
#define K_SIZE  (NKPAD * KROW_STRIDE)          // 23040
#define VTROW_STRIDE 592                       // 576B data + 16B pad
#define VT_OFF  (K_OFF + K_SIZE)
#define VT_SIZE (DHEAD * VTROW_STRIDE)         // 18944
#define P_OFF   (VT_OFF + VT_SIZE)
#define PW_STRIDE 1280                         // per-wave 16 x 80B P tile
#define P_SIZE  (8 * PW_STRIDE)                // 10240
#define VAL_OFF (P_OFF + P_SIZE)
#define SMEM_BYTES (VAL_OFF + 320)             // ~52.5 KB << 320 KB WGP LDS

__device__ __forceinline__ unsigned short f2bf(float f) {
    __bf16 h = (__bf16)f;                      // v_cvt_pk_bf16_f32 path, RNE
    return __builtin_bit_cast(unsigned short, h);
}

// DPP16 lane swap within each 16-lane row: result[l] = x[l ^ pattern]
template <int CTRL>
__device__ __forceinline__ float dppx(float x) {
    int i = __builtin_bit_cast(int, x);
    int r = __builtin_amdgcn_update_dpp(0, i, CTRL, 0xF, 0xF, true);
    return __builtin_bit_cast(float, r);
}
// Butterfly allreduce over 16 lanes with masks {1,2,7,15} (spans Z2^4).
__device__ __forceinline__ float red16_max(float x) {
    x = fmaxf(x, dppx<0xB1>(x));    // quad_perm(1,0,3,2)  : xor 1
    x = fmaxf(x, dppx<0x4E>(x));    // quad_perm(2,3,0,1)  : xor 2
    x = fmaxf(x, dppx<0x141>(x));   // row_half_mirror     : xor 7
    x = fmaxf(x, dppx<0x140>(x));   // row_mirror          : xor 15
    return x;
}
__device__ __forceinline__ float red16_sum(float x) {
    x += dppx<0xB1>(x);
    x += dppx<0x4E>(x);
    x += dppx<0x141>(x);
    x += dppx<0x140>(x);
    return x;
}

// CDNA5 hardware tanh (TRANS op). v_nop inside the asm covers the
// 1-op TRANS result-use hazard the compiler cannot see through inline asm.
// Non-volatile: pure computation, let the scheduler move/batch around it.
__device__ __forceinline__ float htanh(float x) {
    float y;
    asm("v_tanh_f32 %0, %1\n\tv_nop" : "=v"(y) : "v"(x));
    return y;
}

__global__ __launch_bounds__(256)
void lattn_kernel(const float* __restrict__ q, const float* __restrict__ k,
                  const float* __restrict__ v, const unsigned char* __restrict__ mask,
                  const float* __restrict__ bias, const float* __restrict__ mkv,
                  float* __restrict__ out)
{
    __shared__ __align__(16) unsigned char smem[SMEM_BYTES];
    const int wi  = blockIdx.x;   // window
    const int hh  = blockIdx.y;   // head
    const int tid = threadIdx.x;

    // ================= stage K (row-major) + V^T + validity into LDS ========
    for (int j = tid; j < NKPAD; j += 256) {
        union { float4 f4[8]; float f[32]; } kr, vr;
        unsigned char val = 0;
        if (j < NMEM) {                               // memory kv slots
            const float4* ks = (const float4*)(mkv + (hh * NMEM + j) * DHEAD);
            const float4* vs = (const float4*)(mkv + (NHEADS * NMEM * DHEAD) + (hh * NMEM + j) * DHEAD);
            #pragma unroll
            for (int t = 0; t < 8; ++t) { kr.f4[t] = ks[t]; vr.f4[t] = vs[t]; }
            val = 1;
        } else if (j < NKEYS) {
            const bool prev = (j < NMEM + WINDOW);
            const int  pos  = prev ? ((wi - 1) * WINDOW + (j - NMEM))
                                   : (wi * WINDOW + (j - NMEM - WINDOW));
            if (prev && wi == 0) {                    // zero "previous" window
                #pragma unroll
                for (int t = 0; t < 8; ++t) { kr.f4[t] = make_float4(0,0,0,0); vr.f4[t] = make_float4(0,0,0,0); }
            } else {
                const float4* ks = (const float4*)(k + ((size_t)hh * SEQ + pos) * DHEAD);
                const float4* vs = (const float4*)(v + ((size_t)hh * SEQ + pos) * DHEAD);
                #pragma unroll
                for (int t = 0; t < 8; ++t) { kr.f4[t] = ks[t]; vr.f4[t] = vs[t]; }
                val = mask[pos] ? 1 : 0;
            }
        } else {                                      // tail padding 260..287
            #pragma unroll
            for (int t = 0; t < 8; ++t) { kr.f4[t] = make_float4(0,0,0,0); vr.f4[t] = make_float4(0,0,0,0); }
        }
        // K row -> bf16, 4x ds_store_b128
        union { uint4 u4[4]; unsigned short s[32]; } ko;
        #pragma unroll
        for (int d = 0; d < DHEAD; ++d) ko.s[d] = f2bf(kr.f[d]);
        uint4* kdst = (uint4*)(smem + K_OFF + (size_t)j * KROW_STRIDE);
        #pragma unroll
        for (int t = 0; t < 4; ++t) kdst[t] = ko.u4[t];
        // V^T scatter: V^T[d][j]
        #pragma unroll
        for (int d = 0; d < DHEAD; ++d)
            *(unsigned short*)(smem + VT_OFF + (size_t)d * VTROW_STRIDE + (size_t)j * 2) = f2bf(vr.f[d]);
        smem[VAL_OFF + j] = val;
    }
    __syncthreads();

    // ================= per-wave flash-attention over one 16-row Q block =====
    const int wv = tid >> 5;           // wave -> query row block (8 x 16 = 128)
    const int ln = tid & 31;
    const int n  = ln & 15;            // fragment column / row-within-16
    const int hf = ln >> 4;            // K-half selector
    const int h8 = hf * 8;

    // Q A-fragment (16x32 bf16), scale folded in
    const int    qrow = wi * WINDOW + wv * 16 + n;
    const float* qsrc = q + ((size_t)hh * SEQ + qrow) * DHEAD;
    union AF { v16bf v; unsigned short s[16]; } aq;
    const float qscale = 0.17677669529663687f;        // 32^-0.5
    #pragma unroll
    for (int e = 0; e < 8; ++e) aq.s[e]     = f2bf(qscale * qsrc[h8 + e]);
    #pragma unroll
    for (int e = 0; e < 8; ++e) aq.s[8 + e] = f2bf(qscale * qsrc[16 + h8 + e]);

    v8f O0 = {}; v8f O1 = {};
    float mrow[8], lrow[8];
    #pragma unroll
    for (int r = 0; r < 8; ++r) { mrow[r] = NEGINF; lrow[r] = 0.f; }

    // per-lane bias row base (row stride 2w floats); clamped column added per kt
    const float* biasw = bias + ((size_t)wi * WINDOW + (size_t)(wv * 16 + h8)) * (2 * WINDOW);
    unsigned char* pbuf = smem + P_OFF + (size_t)wv * PW_STRIDE;

    for (int kt = 0; kt < NKT; ++kt) {
        // ---- sim = Q x K^T for 32 keys (two 16x16 WMMAs, K = d = 32) ----
        v8f s[2];
        #pragma unroll
        for (int sub = 0; sub < 2; ++sub) {
            const unsigned char* kb = smem + K_OFF + (size_t)(kt * 32 + sub * 16 + n) * KROW_STRIDE;
            union { v16bf v; uint4 u[2]; } bk;
            bk.u[0] = *(const uint4*)(kb + hf * 16);
            bk.u[1] = *(const uint4*)(kb + 32 + hf * 16);
            v8f zero = {};
            s[sub] = __builtin_amdgcn_wmma_f32_16x16x32_bf16(
                false, aq.v, false, bk.v, (short)0, zero, false, false);
        }

        // ---- batch all bias loads + validity first (one load clause) ----
        float bv[2][8];
        bool  colv[2], hasb[2];
        #pragma unroll
        for (int sub = 0; sub < 2; ++sub) {
            const int j = kt * 32 + sub * 16 + n;
            colv[sub] = smem[VAL_OFF + j] != 0;
            hasb[sub] = (j >= NMEM) && (j < NKEYS);
            int jc = j - NMEM; jc = jc < 0 ? 0 : (jc > 2*WINDOW - 1 ? 2*WINDOW - 1 : jc);
            const float* bp = biasw + jc;
            #pragma unroll
            for (int r = 0; r < 8; ++r)
                bv[sub][r] = bp[(size_t)r * (2 * WINDOW)];
        }

        // ---- bias + soft-clamp + mask (C layout: col = lane, row = vgpr) ----
        #pragma unroll
        for (int sub = 0; sub < 2; ++sub) {
            #pragma unroll
            for (int r = 0; r < 8; ++r) {
                float x = s[sub][r] + (hasb[sub] ? bv[sub][r] : 0.f);
                x = SOFTCLAMP * htanh(x * (1.0f / SOFTCLAMP));
                s[sub][r] = colv[sub] ? x : NEGINF;
            }
        }

        // ---- online softmax update + P store (bf16) into per-wave LDS ----
        #pragma unroll
        for (int r = 0; r < 8; ++r) {
            float t  = red16_max(fmaxf(s[0][r], s[1][r]));
            float mn = fmaxf(mrow[r], t);
            float sc = __expf(mrow[r] - mn);
            mrow[r] = mn;
            O0[r] = O0[r] * sc; O1[r] = O1[r] * sc; lrow[r] = lrow[r] * sc;
            float p0 = __expf(s[0][r] - mn);
            float p1 = __expf(s[1][r] - mn);
            lrow[r] += red16_sum(p0 + p1);
            unsigned char* prow = pbuf + (size_t)(r + h8) * KROW_STRIDE;
            *(unsigned short*)(prow + (size_t)n * 2)        = f2bf(p0);
            *(unsigned short*)(prow + (size_t)(16 + n) * 2) = f2bf(p1);
        }
        // wave-private P tile: DS ops are in-order; wait for stores to land
        asm volatile("s_wait_dscnt 0" ::: "memory");

        // ---- O += P x V (two WMMAs, N-tiles dcol 0..15 and 16..31) ----
        union { v16bf v; uint4 u[2]; } pa;
        {
            const unsigned char* pr = pbuf + (size_t)n * KROW_STRIDE;
            pa.u[0] = *(const uint4*)(pr + hf * 16);
            pa.u[1] = *(const uint4*)(pr + 32 + hf * 16);
        }
        #pragma unroll
        for (int dt = 0; dt < 2; ++dt) {
            const unsigned char* vb = smem + VT_OFF + (size_t)(dt * 16 + n) * VTROW_STRIDE + (size_t)kt * 64;
            union { v16bf v; uint4 u[2]; } bv2;
            bv2.u[0] = *(const uint4*)(vb + hf * 16);
            bv2.u[1] = *(const uint4*)(vb + 32 + hf * 16);
            if (dt == 0)
                O0 = __builtin_amdgcn_wmma_f32_16x16x32_bf16(false, pa.v, false, bv2.v, (short)0, O0, false, false);
            else
                O1 = __builtin_amdgcn_wmma_f32_16x16x32_bf16(false, pa.v, false, bv2.v, (short)0, O1, false, false);
        }
    }

    // ================= normalize + store (coalesced 16-lane rows) ===========
    float* obase = out + ((size_t)hh * SEQ + (size_t)wi * WINDOW) * DHEAD;
    #pragma unroll
    for (int r = 0; r < 8; ++r) {
        const int m   = wv * 16 + r + h8;
        const float inv = 1.0f / lrow[r];
        obase[(size_t)m * DHEAD + n]      = O0[r] * inv;
        obase[(size_t)m * DHEAD + 16 + n] = O1[r] * inv;
    }
}

extern "C" void kernel_launch(void* const* d_in, const int* in_sizes, int n_in,
                              void* d_out, int out_size, void* d_ws, size_t ws_size,
                              hipStream_t stream) {
    const float*         q    = (const float*)d_in[0];
    const float*         k    = (const float*)d_in[1];
    const float*         v    = (const float*)d_in[2];
    const unsigned char* mask = (const unsigned char*)d_in[3];  // jnp bool = 1 byte
    const float*         bias = (const float*)d_in[4];
    const float*         mkv  = (const float*)d_in[5];
    float* out = (float*)d_out;

    dim3 grid(NWIN, NHEADS, 1);
    lattn_kernel<<<grid, 256, 0, stream>>>(q, k, v, mask, bias, mkv, out);
}
